// Attention_38680475468315
// MI455X (gfx1250) — matrix operand order repaired and checked by reading, not compiled, not run on hardware
//
#include <hip/hip_runtime.h>

// ---------------- problem constants ----------------
#define BB   4096            // batch
#define TT   26              // time steps
#define INF  512             // encoder feature dim
#define HH   512             // hidden
#define GG   2048            // 4*H gates
#define NCC  97              // classes
#define NEE  256             // embedding dim
#define NVOC 98              // NC+1 vocab rows

#define LDS_PITCH 520        // u16 per staged row: 512 + 8 pad -> bank-conflict-free
#define BLK 128              // 4 waves per block for the LDS-staged GEMMs

#define AS3 __attribute__((address_space(3)))

// ---------------- types ----------------
typedef unsigned short u16;
typedef __bf16 bf16_t;
typedef bf16_t bfx16 __attribute__((ext_vector_type(16)));
typedef float  f32x8 __attribute__((ext_vector_type(8)));
typedef float  f32x4 __attribute__((ext_vector_type(4)));
typedef unsigned int u32x4 __attribute__((ext_vector_type(4)));
typedef unsigned short u16x4 __attribute__((ext_vector_type(4)));
typedef unsigned int u32x4v __attribute__((ext_vector_type(4)));
typedef int i32x4v __attribute__((ext_vector_type(4)));
typedef int i32x8v __attribute__((ext_vector_type(8)));

typedef AS3 u16   ls_u16;    // LDS-resident halfword
typedef AS3 u32x4 ls_u32x4;  // LDS-resident 128-bit chunk

union U16x16 { bfx16 v; u32x4 q[2]; };

// LDS byte offset of a __shared__ object = low 32 bits of its flat address
// (ISA 10.2: LDS aperture maps addr[31:0] to the LDS offset).
__device__ inline unsigned lds_off_of(const void* p) {
    return (unsigned)(size_t)p;
}
__device__ inline const ls_u16* lds_cptr(const void* p) {
    return (const ls_u16*)(size_t)lds_off_of(p);
}
__device__ inline ls_u16* lds_ptr(void* p) {
    return (ls_u16*)(size_t)lds_off_of(p);
}

__device__ inline u16 f2bf(float f) {           // round-to-nearest-even
    unsigned u = __float_as_uint(f);
    unsigned r = u + 0x7FFFu + ((u >> 16) & 1u);
    return (u16)(r >> 16);
}

__device__ inline bfx16 ld16(const u16* p0, const u16* p1) {
    U16x16 u;
    u.q[0] = *reinterpret_cast<const u32x4*>(p0);
    u.q[1] = *reinterpret_cast<const u32x4*>(p1);
    return u.v;
}

// A-matrix (16-bit, 16x32): lane L holds row M=L&15; K = 16*(L>>4) .. +15
// contiguous (matches B fragment order)
__device__ inline bfx16 ld_a(const u16* arow, int k0, int half) {
    const u16* p = arow + k0 + half * 16;
    return ld16(p, p + 8);
}

// B-matrix (16-bit, 32x16 KxN) from row-major [N][ldk] global memory:
// lane L holds column N = nbase + (L&15), K = 16*(L>>4) .. +15 contiguous
__device__ inline bfx16 ld_b(const u16* W, int ldk, int nbase, int k0, int lane) {
    const u16* p = W + (size_t)(nbase + (lane & 15)) * ldk + k0 + ((lane >> 4) << 4);
    return ld16(p, p + 8);
}

// Same fragment from an LDS-staged tile (pitch LDS_PITCH) -> ds_load_b128 pair
__device__ inline bfx16 ld_b_lds(const ls_u16* lw, int nbase, int k0, int lane) {
    const ls_u16* p = lw + (nbase + (lane & 15)) * LDS_PITCH + k0 + ((lane >> 4) << 4);
    U16x16 u;
    u.q[0] = *(const ls_u32x4*)p;
    u.q[1] = *(const ls_u32x4*)(p + 8);
    return u.v;
}

__device__ inline f32x8 wmma_bf16(bfx16 a, bfx16 b, f32x8 c) {
    return __builtin_amdgcn_wmma_f32_16x16x32_bf16(
        false, a, false, b, (short)0, c, false, false);
}

__device__ inline float sigm(float x) { return 1.0f / (1.0f + __expf(-x)); }

// ---------------- Tensor Data Mover staging ----------------
#if __has_builtin(__builtin_amdgcn_tensor_load_to_lds)
#define HAVE_TDM 1
#if __has_include(<hip/amd_detail/amd_gfx1250_TDM.h>)
#define TDM_SIX_ARG 1
#else
#define TDM_SIX_ARG 0
#endif
__device__ inline void tdm_raw(u32x4v g0, i32x8v g1) {
    i32x4v z4 = {};
#if TDM_SIX_ARG
    i32x8v z8 = {};
    __builtin_amdgcn_tensor_load_to_lds(g0, g1, z4, z4, z8, 0);
#else
    __builtin_amdgcn_tensor_load_to_lds(g0, g1, z4, z4, 0);
#endif
}

// Stage `rows` rows of `rowlen` bf16 (src row stride `sstride` elems) into LDS
// at byte offset lds_off, with HW padding of 4 DWORDs every 256 DWORDs so the
// effective LDS pitch is LDS_PITCH (= 512+8) halfwords.
__device__ inline void tdm_stage_rows(const u16* gsrc, unsigned lds_off,
                                      int rows, int rowlen, int sstride) {
    unsigned long long ga = (unsigned long long)(uintptr_t)gsrc;
    u32x4v g0 = {};
    i32x8v g1 = {};
    g0[0] = 1u;                                              // count=1
    g0[1] = lds_off;                                         // lds_addr
    g0[2] = (unsigned)ga;                                    // global_addr lo
    g0[3] = (unsigned)((ga >> 32) & 0x1FFFFFFu) | (2u << 30);// addr hi | type=2
    g1[0] = (1 << 16)        // data_size = 2 bytes
          | (1 << 20)        // pad_enable
          | (7 << 22)        // pad_interval = 256 DWORDs (one 512-elem row)
          | (3 << 25);       // pad_amount   = 4 DWORDs (8 halfwords)
    g1[1] = (rowlen & 0xFFFF) << 16;                         // tensor_dim0 lo16
    g1[2] = ((rowlen >> 16) & 0xFFFF) | ((rows & 0xFFFF) << 16); // dim0 hi | dim1 lo
    g1[3] = ((rows >> 16) & 0xFFFF) | ((rowlen & 0xFFFF) << 16); // dim1 hi | tile_dim0
    g1[4] = rows & 0xFFFF;                                   // tile_dim1 (tile_dim2=0)
    g1[5] = sstride;                                         // tensor_dim0_stride lo32
    g1[6] = 0; g1[7] = 0;
    tdm_raw(g0, g1);
}
#else
#define HAVE_TDM 0
#endif

// ---------------- utility kernels ----------------
__global__ void k_cvt_bf16(const float* __restrict__ s, u16* __restrict__ d, long n4) {
    long i = (long)blockIdx.x * blockDim.x + threadIdx.x;
    if (i < n4) {
        f32x4 f = *reinterpret_cast<const f32x4*>(s + 4 * i);
        u16x4 o;
        o[0] = f2bf(f[0]); o[1] = f2bf(f[1]); o[2] = f2bf(f[2]); o[3] = f2bf(f[3]);
        *reinterpret_cast<u16x4*>(d + 4 * i) = o;
    }
}

__global__ void k_zero_u32(unsigned* __restrict__ p, long n) {
    long i = (long)blockIdx.x * blockDim.x + threadIdx.x;
    if (i < n) p[i] = 0u;
}

// ---------------- emb_table @ W_ih[:,512:]^T  -> [98, 2048] ----------------
__global__ __launch_bounds__(32) void k_emb_gemm(
    const u16* __restrict__ Eb,     // [98,256] bf16
    const u16* __restrict__ Wih,    // [2048,768] bf16
    float* __restrict__ embgx)      // [98,2048] f32
{
    const int lane = threadIdx.x;
    const int mt = blockIdx.x, nt = blockIdx.y;
    const int n0 = nt * 64;
    const int half = lane >> 4, ml = lane & 15;

    int ra = mt * 16 + ml; if (ra > NVOC - 1) ra = NVOC - 1;   // clamp A reads
    const u16* arow = Eb + (size_t)ra * NEE;

    f32x8 a0 = {}, a1 = {}, a2 = {}, a3 = {};
#pragma unroll
    for (int k0 = 0; k0 < NEE; k0 += 32) {
        bfx16 a  = ld_a(arow, k0, half);
        bfx16 b0 = ld_b(Wih, INF + NEE, n0,      INF + k0, lane);
        bfx16 b1 = ld_b(Wih, INF + NEE, n0 + 16, INF + k0, lane);
        bfx16 b2 = ld_b(Wih, INF + NEE, n0 + 32, INF + k0, lane);
        bfx16 b3 = ld_b(Wih, INF + NEE, n0 + 48, INF + k0, lane);
        a0 = wmma_bf16(a, b0, a0); a1 = wmma_bf16(a, b1, a1);
        a2 = wmma_bf16(a, b2, a2); a3 = wmma_bf16(a, b3, a3);
    }
#pragma unroll
    for (int e = 0; e < 8; e++) {
        int mg = mt * 16 + e + half * 8;
        if (mg < NVOC) {
            float* o = embgx + (size_t)mg * GG + n0 + ml;
            o[0] = a0[e]; o[16] = a1[e]; o[32] = a2[e]; o[48] = a3[e];
        }
    }
}

// ------- gx = batch_H @ W_ih[:,:512]^T + b_ih + embgx[text]  -> [T,B,2048] -------
// 4 waves/block share one 64-col W_ih strip staged in LDS (TDM); each wave owns
// a 16-row batch tile; A prefetched one K-step ahead, B via ds_load_b128.
__global__ __launch_bounds__(BLK) void k_gemm_gx(
    const u16* __restrict__ Xb,     // [B*T, 512] bf16 (row = b*T+t)
    const u16* __restrict__ Wih,    // [2048,768] bf16
    const float* __restrict__ b_ih, // [2048]
    const float* __restrict__ embgx,// [98,2048]
    const int* __restrict__ text,   // [B,T]
    float* __restrict__ gx)         // [T,B,2048]
{
    __shared__ __attribute__((aligned(16))) u16 lB[64 * LDS_PITCH];
    const int tid = threadIdx.x;
    const int lane = tid & 31, wid = tid >> 5;
    const int mt = blockIdx.x, nt = blockIdx.y;
    const int n0 = nt * 64;
    const int half = lane >> 4, ml = lane & 15;
    const unsigned lbase = lds_off_of(lB);

    // ---- stage W_ih[n0:n0+64, 0:512] into LDS ----
#if HAVE_TDM
    if (tid < 32) {                      // wave 0 issues the DMA
        tdm_stage_rows(Wih + (size_t)n0 * (INF + NEE), lbase, 64, INF, INF + NEE);
        __builtin_amdgcn_s_wait_tensorcnt(0);
    }
#else
    {
        ls_u16* lz = lds_ptr(lB);
        for (int i = tid; i < 64 * (INF / 8); i += BLK) {
            int r = i / (INF / 8), c = (i % (INF / 8)) * 8;
            u32x4 v = *reinterpret_cast<const u32x4*>(
                Wih + (size_t)(n0 + r) * (INF + NEE) + c);
            *(ls_u32x4*)(lz + r * LDS_PITCH + c) = v;   // ds_store_b128
        }
    }
#endif
    __syncthreads();

    const int mrow = mt * 64 + wid * 16;
    const u16* arow = Xb + (size_t)(mrow + ml) * INF;
    const ls_u16* lw = lds_cptr(lB);

    f32x8 a0 = {}, a1 = {}, a2 = {}, a3 = {};
    bfx16 a = ld_a(arow, 0, half);
#pragma unroll
    for (int k0 = 0; k0 < INF; k0 += 32) {
        bfx16 an = a;
        if (k0 + 32 < INF) an = ld_a(arow, k0 + 32, half);  // prefetch next A
        bfx16 b0 = ld_b_lds(lw, 0,  k0, lane);
        bfx16 b1 = ld_b_lds(lw, 16, k0, lane);
        bfx16 b2 = ld_b_lds(lw, 32, k0, lane);
        bfx16 b3 = ld_b_lds(lw, 48, k0, lane);
        a0 = wmma_bf16(a, b0, a0); a1 = wmma_bf16(a, b1, a1);
        a2 = wmma_bf16(a, b2, a2); a3 = wmma_bf16(a, b3, a3);
        a = an;
    }
    const int nl = n0 + ml;
    const float bi0 = b_ih[nl], bi1 = b_ih[nl + 16], bi2 = b_ih[nl + 32], bi3 = b_ih[nl + 48];
#pragma unroll
    for (int e = 0; e < 8; e++) {
        int r = mrow + e + half * 8;               // = b*T + t
        int b = r / TT, t = r % TT;
        int cidx = text[r];
        const float* eg = embgx + (size_t)cidx * GG + nl;
        float* o = gx + ((size_t)t * BB + b) * GG + nl;
        o[0]  = a0[e] + bi0 + eg[0];
        o[16] = a1[e] + bi1 + eg[16];
        o[32] = a2[e] + bi2 + eg[32];
        o[48] = a3[e] + bi3 + eg[48];
    }
}

// ------- one LSTM step: hh-GEMM (W_hh staged via TDM) + gates + cell update -------
// 4 waves/block cover 64 batch rows; the 4 gate tiles (16 cols each, all K)
// are staged in LDS once per block.
__global__ __launch_bounds__(BLK) void k_lstm_step(
    const float* __restrict__ gx_t,  // [B,2048]
    const u16* __restrict__ hprev,   // [B,512] bf16
    const u16* __restrict__ Whh,     // [2048,512] bf16
    const float* __restrict__ b_hh,  // [2048]
    float* __restrict__ cbuf,        // [B,512] f32 in/out
    u16* __restrict__ hnew,          // [B,512] bf16 (hs slice for step t)
    float* __restrict__ out_h,       // [B,T,512] f32 (d_out hiddens)
    int t)
{
    __shared__ __attribute__((aligned(16))) u16 lW[64 * LDS_PITCH];
    const int tid = threadIdx.x;
    const int lane = tid & 31, wid = tid >> 5;
    const int bt = blockIdx.x, nt = blockIdx.y;
    const int n0 = nt * 16;
    const int half = lane >> 4, ml = lane & 15;
    const unsigned lbase = lds_off_of(lW);

    // ---- stage W_hh[gi*512+n0 : +16, 0:512] for gi=0..3 into LDS ----
#if HAVE_TDM
    if (tid < 32) {
#pragma unroll
        for (int gi = 0; gi < 4; gi++)
            tdm_stage_rows(Whh + (size_t)(gi * HH + n0) * HH,
                           lbase + (unsigned)(gi * 16 * LDS_PITCH * 2), 16, HH, HH);
        __builtin_amdgcn_s_wait_tensorcnt(0);
    }
#else
    {
        ls_u16* lz = lds_ptr(lW);
        for (int i = tid; i < 64 * (HH / 8); i += BLK) {
            int r = i / (HH / 8), c = (i % (HH / 8)) * 8;
            int gi = r >> 4, nr = r & 15;
            u32x4 v = *reinterpret_cast<const u32x4*>(
                Whh + (size_t)(gi * HH + n0 + nr) * HH + c);
            *(ls_u32x4*)(lz + r * LDS_PITCH + c) = v;   // ds_store_b128
        }
    }
#endif
    __syncthreads();

    const int rb = bt * 64 + wid * 16;
    const u16* arow = hprev + (size_t)(rb + ml) * HH;
    const ls_u16* lw = lds_cptr(lW);

    f32x8 ai = {}, af = {}, ag = {}, ao = {};
    bfx16 a = ld_a(arow, 0, half);
#pragma unroll
    for (int k0 = 0; k0 < HH; k0 += 32) {
        bfx16 an = a;
        if (k0 + 32 < HH) an = ld_a(arow, k0 + 32, half);   // prefetch next A
        bfx16 wi = ld_b_lds(lw, 0,  k0, lane);
        bfx16 wf = ld_b_lds(lw, 16, k0, lane);
        bfx16 wg = ld_b_lds(lw, 32, k0, lane);
        bfx16 wo = ld_b_lds(lw, 48, k0, lane);
        ai = wmma_bf16(a, wi, ai); af = wmma_bf16(a, wf, af);
        ag = wmma_bf16(a, wg, ag); ao = wmma_bf16(a, wo, ao);
        a = an;
    }
    const int n = n0 + ml;
    const float bhi = b_hh[n], bhf = b_hh[HH + n], bhg = b_hh[2 * HH + n], bho = b_hh[3 * HH + n];
#pragma unroll
    for (int e = 0; e < 8; e++) {
        int b = rb + e + half * 8;
        size_t g = (size_t)b * GG + n;
        float gi = sigm (ai[e] + gx_t[g]          + bhi);
        float gf = sigm (af[e] + gx_t[g + HH]     + bhf);
        float gg = tanhf(ag[e] + gx_t[g + 2 * HH] + bhg);
        float go = sigm (ao[e] + gx_t[g + 3 * HH] + bho);
        size_t ci = (size_t)b * HH + n;
        float cn = gf * cbuf[ci] + gi * gg;
        cbuf[ci] = cn;
        float hn = go * tanhf(cn);
        hnew[ci] = f2bf(hn);
        out_h[((size_t)b * TT + t) * HH + n] = hn;
    }
}

// ------- probs = hs @ W_gen^T + b_gen  -> [B,T,97] -------
// 4 waves/block, all 112 padded W_gen rows staged in LDS once per block.
__global__ __launch_bounds__(BLK) void k_gemm_probs(
    const u16* __restrict__ Hs,     // [T*B, 512] bf16 (row = t*B+b)
    const u16* __restrict__ Wg,     // [112,512] bf16 (rows >=97 junk, unstored)
    const float* __restrict__ b_gen,// [97]
    float* __restrict__ probs)      // [B,T,97]
{
    __shared__ __attribute__((aligned(16))) u16 lG[112 * LDS_PITCH];
    const int tid = threadIdx.x;
    const int lane = tid & 31, wid = tid >> 5;
    const int mt = blockIdx.x;
    const int half = lane >> 4, ml = lane & 15;
    const unsigned lbase = lds_off_of(lG);

#if HAVE_TDM
    if (tid < 32) {
        tdm_stage_rows(Wg, lbase, 112, HH, HH);
        __builtin_amdgcn_s_wait_tensorcnt(0);
    }
#else
    {
        ls_u16* lz = lds_ptr(lG);
        for (int i = tid; i < 112 * (HH / 8); i += BLK) {
            int r = i / (HH / 8), c = (i % (HH / 8)) * 8;
            u32x4 v = *reinterpret_cast<const u32x4*>(Wg + (size_t)r * HH + c);
            *(ls_u32x4*)(lz + r * LDS_PITCH + c) = v;
        }
    }
#endif
    __syncthreads();

    const int mrow = mt * 64 + wid * 16;
    const u16* arow = Hs + (size_t)(mrow + ml) * HH;
    const ls_u16* lw = lds_cptr(lG);

    f32x8 acc[7];
#pragma unroll
    for (int j = 0; j < 7; j++) acc[j] = (f32x8){};
    bfx16 a = ld_a(arow, 0, half);
#pragma unroll 2
    for (int k0 = 0; k0 < HH; k0 += 32) {
        bfx16 an = a;
        if (k0 + 32 < HH) an = ld_a(arow, k0 + 32, half);   // prefetch next A
#pragma unroll
        for (int j = 0; j < 7; j++) {
            bfx16 b = ld_b_lds(lw, j * 16, k0, lane);
            acc[j] = wmma_bf16(a, b, acc[j]);
        }
        a = an;
    }
#pragma unroll
    for (int e = 0; e < 8; e++) {
        int r = mrow + e + half * 8;            // = t*B + b  (B = 4096 = 2^12)
        int t = r >> 12, b = r & (BB - 1);
        float* orow = probs + ((size_t)b * TT + t) * NCC;
#pragma unroll
        for (int j = 0; j < 7; j++) {
            int cc = j * 16 + ml;
            if (cc < NCC) orow[cc] = acc[j][e] + b_gen[cc];
        }
    }
}

// ---------------- host side ----------------
static inline long cdiv(long a, long b) { return (a + b - 1) / b; }

extern "C" void kernel_launch(void* const* d_in, const int* in_sizes, int n_in,
                              void* d_out, int out_size, void* d_ws, size_t ws_size,
                              hipStream_t stream) {
    (void)in_sizes; (void)n_in; (void)out_size; (void)ws_size;
    const float* batch_H   = (const float*)d_in[0];
    const int*   text      = (const int*)  d_in[1];
    const float* emb_table = (const float*)d_in[2];
    const float* W_ih      = (const float*)d_in[3];
    const float* W_hh      = (const float*)d_in[4];
    const float* b_ih      = (const float*)d_in[5];
    const float* b_hh      = (const float*)d_in[6];
    const float* W_gen     = (const float*)d_in[7];
    const float* b_gen     = (const float*)d_in[8];

    float* probs = (float*)d_out;                       // [B,T,97]
    float* out_h = probs + (size_t)BB * TT * NCC;       // [B,T,512]

    // workspace carve-up (all chunks 256B aligned)
    char* ws = (char*)d_ws;
    size_t off = 0;
    u16*   xbf   = (u16*)  (ws + off); off += (size_t)BB * TT * INF * 2;      // 109 MB
    u16*   wih   = (u16*)  (ws + off); off += (size_t)GG * (INF + NEE) * 2;   // 3 MB
    u16*   whh   = (u16*)  (ws + off); off += (size_t)GG * HH * 2;            // 2 MB
    u16*   wgen  = (u16*)  (ws + off); off += (size_t)112 * HH * 2;           // padded rows
    u16*   ebf   = (u16*)  (ws + off); off += (size_t)NVOC * NEE * 2;
    float* embgx = (float*)(ws + off); off += (size_t)NVOC * GG * 4;
    float* gx    = (float*)(ws + off); off += (size_t)TT * BB * GG * 4;       // 872 MB
    u16*   hs    = (u16*)  (ws + off); off += (size_t)TT * BB * HH * 2;       // 109 MB
    u16*   h0    = (u16*)  (ws + off); off += (size_t)BB * HH * 2;
    float* cbuf  = (float*)(ws + off); off += (size_t)BB * HH * 4;

    // 1) fp32 -> bf16 converts (vectorized x4)
    {
        long n4;
        n4 = (long)BB * TT * INF / 4;
        k_cvt_bf16<<<dim3(cdiv(n4, 256)), 256, 0, stream>>>(batch_H, xbf, n4);
        n4 = (long)GG * (INF + NEE) / 4;
        k_cvt_bf16<<<dim3(cdiv(n4, 256)), 256, 0, stream>>>(W_ih, wih, n4);
        n4 = (long)GG * HH / 4;
        k_cvt_bf16<<<dim3(cdiv(n4, 256)), 256, 0, stream>>>(W_hh, whh, n4);
        n4 = (long)NCC * HH / 4;
        k_cvt_bf16<<<dim3(cdiv(n4, 256)), 256, 0, stream>>>(W_gen, wgen, n4);
        n4 = (long)NVOC * NEE / 4;
        k_cvt_bf16<<<dim3(cdiv(n4, 256)), 256, 0, stream>>>(emb_table, ebf, n4);
    }
    // 2) zero h0 and c (re-zeroed every call: deterministic)
    {
        long n = (long)BB * HH / 2;   // h0: B*H bf16 = B*H/2 u32
        k_zero_u32<<<dim3(cdiv(n, 256)), 256, 0, stream>>>((unsigned*)h0, n);
        n = (long)BB * HH;            // c: B*H f32
        k_zero_u32<<<dim3(cdiv(n, 256)), 256, 0, stream>>>((unsigned*)cbuf, n);
    }
    // 3) embedding-projection table [98,2048]
    k_emb_gemm<<<dim3(7, GG / 64), 32, 0, stream>>>(ebf, wih, embgx);

    // 4) big input-projection GEMM -> gx[T,B,2048]
    k_gemm_gx<<<dim3((BB * TT) / 64, GG / 64), BLK, 0, stream>>>(
        xbf, wih, b_ih, embgx, text, gx);

    // 5) 26 fused recurrent steps (stream-ordered dependency chain)
    for (int t = 0; t < TT; t++) {
        const u16* hp = (t == 0) ? h0 : hs + (size_t)(t - 1) * BB * HH;
        k_lstm_step<<<dim3(BB / 64, HH / 16), BLK, 0, stream>>>(
            gx + (size_t)t * BB * GG, hp, whh, b_hh, cbuf,
            hs + (size_t)t * BB * HH, out_h, t);
    }

    // 6) readout GEMM (4 waves/block, W_gen in LDS)
    k_gemm_probs<<<dim3((TT * BB) / 64), BLK, 0, stream>>>(hs, wgen, b_gen, probs);
}
